// AbstractRecurrent_64845416235064
// MI455X (gfx1250) — matrix operand order repaired and checked by reading, not compiled, not run on hardware
//
#include <hip/hip_runtime.h>

// Problem constants (from reference): N=32, C_in=C_out=512, T=1024
#define TSTEPS 1024
#define NB     32
#define CIN    512
#define COUT   512

#define CUR_KEEP  0.75f   // 1 - CURRENT_DECAY
#define VOLT_KEEP 0.90f   // 1 - VOLTAGE_DECAY
#define THRESH    1.0f

#define SCAN_WGS 8        // persistent workgroups in the scan kernel

typedef __attribute__((ext_vector_type(16))) __bf16 v16bf;
typedef __attribute__((ext_vector_type(8)))  float  v8f;

union BF16x16 { v16bf v; __bf16 h[16]; unsigned short u[16]; };

// ---------------------------------------------------------------------------
// init: zero double-buffered spike buffer + global barrier counter (per call)
// ---------------------------------------------------------------------------
__global__ void lif_init(unsigned short* __restrict__ sbuf,
                         unsigned* __restrict__ counter) {
    int i = blockIdx.x * blockDim.x + threadIdx.x;
    if (i < 2 * NB * COUT) sbuf[i] = 0;
    if (i == 0) *counter = 0;
}

// Convert 16 f32 (from 4 float4) to a bf16 B-fragment using HW converts.
__device__ __forceinline__ v16bf cvt_frag(float4 a, float4 b, float4 c, float4 d) {
    BF16x16 r;
    r.h[0]  = (__bf16)a.x; r.h[1]  = (__bf16)a.y; r.h[2]  = (__bf16)a.z; r.h[3]  = (__bf16)a.w;
    r.h[4]  = (__bf16)b.x; r.h[5]  = (__bf16)b.y; r.h[6]  = (__bf16)b.z; r.h[7]  = (__bf16)b.w;
    r.h[8]  = (__bf16)c.x; r.h[9]  = (__bf16)c.y; r.h[10] = (__bf16)c.z; r.h[11] = (__bf16)c.w;
    r.h[12] = (__bf16)d.x; r.h[13] = (__bf16)d.y; r.h[14] = (__bf16)d.z; r.h[15] = (__bf16)d.w;
    return r.v;
}

// ---------------------------------------------------------------------------
// Phase 1: z[t][n][o] = sum_c x[n][c][t] * W[o][c]     (WMMA bf16, f32 acc)
// One wave per 16x16 tile: M = t (16 consecutive t), N = o, fixed n.
// Grid: 64 t-tiles * 32 n * 32 o-tiles = 65536 tiles = 8192 blocks * 8 waves.
// ---------------------------------------------------------------------------
__global__ void input_gemm(const float* __restrict__ x,
                           const float* __restrict__ W,
                           float* __restrict__ z) {
    const int lane  = threadIdx.x & 31;
    const int wid   = blockIdx.x * (blockDim.x >> 5) + (threadIdx.x >> 5);
    const int otile = wid & 31;
    const int n     = (wid >> 5) & 31;
    const int ttile = wid >> 10;
    const int t0 = ttile * 16, o0 = otile * 16;
    const int m = lane & 15, half = lane >> 4, col = lane & 15;

    const float* xcol = x + (size_t)n * CIN * TSTEPS + (t0 + m); // stride TSTEPS per c
    const float* wrow = W + (size_t)(o0 + col) * CIN + half * 16;

    v8f acc = {};
    #pragma unroll 2
    for (int kc = 0; kc < CIN / 32; ++kc) {
        const int c0 = kc * 32;
        // A fragment 16x32 bf16: lane row = m; K = 16*half + j (contiguous, matches B)
        BF16x16 a;
        #pragma unroll
        for (int j = 0; j < 8; ++j) {
            const int k1 = c0 + 16 * half + j;       // K = {0..7} or {16..23}
            const int k2 = c0 + 16 * half + 8 + j;   // K = {8..15} or {24..31}
            a.h[j]     = (__bf16)xcol[(size_t)k1 * TSTEPS];
            a.h[j + 8] = (__bf16)xcol[(size_t)k2 * TSTEPS];
        }
        // B fragment 32x16 bf16: lane col, K = 16*half + j (contiguous in W row)
        const float4* wp = (const float4*)(wrow + c0);
        v16bf b = cvt_frag(wp[0], wp[1], wp[2], wp[3]);

        acc = __builtin_amdgcn_wmma_f32_16x16x32_bf16(false, a.v, false, b,
                                                      (short)0, acc, false, false);
    }
    // C/D layout: VGPR r -> row M = r + 8*half, col = lane%16
    #pragma unroll
    for (int r = 0; r < 8; ++r) {
        const int t = t0 + r + 8 * half;
        z[((size_t)t * NB + n) * COUT + o0 + col] = acc[r];
    }
}

// ---------------------------------------------------------------------------
// Phase 2: persistent LIF scan. 8 WGs * 8 waves = 64 waves; one (n,o) 16x16
// tile per wave; state in VGPRs; R fragments pinned in VGPRs; spikes shared
// through double-buffered bf16 buffer in L2; atomic-counter global barrier.
// ---------------------------------------------------------------------------
__global__ void lif_scan(const float* __restrict__ z,
                         const float* __restrict__ R,
                         float* __restrict__ out,
                         unsigned short* __restrict__ sbuf,  // [2][NB][COUT] bf16
                         unsigned* __restrict__ counter) {
    const int lane  = threadIdx.x & 31;
    const int wid   = blockIdx.x * (blockDim.x >> 5) + (threadIdx.x >> 5); // 0..63
    const int otile = wid & 31;      // p-tile
    const int ntile = wid >> 5;      // 0..1
    const int p0 = otile * 16, n0 = ntile * 16;
    const int m = lane & 15, half = lane >> 4, col = lane & 15;
    const int goff = half ? 16 : 0;  // A-fragment K-group base for this lane

    // Pin R fragments in registers: rec[n,p] = sum_o s[n,o]*R[p,o]
    // B[k=o][col=p] = R[p0+col][k]; lane needs K = kc*32 + 16*half + {0..15}
    v16bf rf[16];
    {
        const float* rrow = R + (size_t)(p0 + col) * COUT + half * 16;
        #pragma unroll
        for (int kc = 0; kc < 16; ++kc) {
            const float4* rp = (const float4*)(rrow + kc * 32);
            rf[kc] = cvt_frag(rp[0], rp[1], rp[2], rp[3]);
        }
    }

    v8f iS = {}, vS = {}, sP = {};   // persistent LIF state, accumulator layout

    // delay_shift: out[..., 0] = 0
    #pragma unroll
    for (int r = 0; r < 8; ++r) {
        const int n = n0 + r + 8 * half, o = p0 + col;
        out[((size_t)n * COUT + o) * TSTEPS] = 0.0f;
    }

    for (int t = 0; t < TSTEPS; ++t) {
        // ---- recurrent GEMM: rec = s_{t-1} @ R^T (16x16 tile, K = 512) ----
        v8f rec = {};
        if (t > 0) {
            const unsigned short* srd =
                sbuf + (size_t)(t & 1) * NB * COUT + (size_t)(n0 + m) * COUT;
            #pragma unroll
            for (int kc = 0; kc < 16; ++kc) {
                const int g = kc * 32 + goff;
                BF16x16 a;
                *(uint4*)&a.u[0] = *(const uint4*)(srd + g);       // K g..g+7
                *(uint4*)&a.u[8] = *(const uint4*)(srd + g + 8);   // K g+8..g+15
                rec = __builtin_amdgcn_wmma_f32_16x16x32_bf16(false, a.v, false, rf[kc],
                                                              (short)0, rec, false, false);
            }
        }

        // ---- z tile for this step (coalesced across lanes 0..15) ----
        const float* zt = z + ((size_t)t * NB + n0) * COUT + p0 + col;
        v8f zv;
        #pragma unroll
        for (int r = 0; r < 8; ++r) zv[r] = zt[(size_t)(r + 8 * half) * COUT];

        // ---- LIF update + stores ----
        unsigned short* swr = sbuf + (size_t)((t + 1) & 1) * NB * COUT;
        #pragma unroll
        for (int r = 0; r < 8; ++r) {
            float i = CUR_KEEP * iS[r] + zv[r] + rec[r];
            float v = VOLT_KEEP * vS[r] * (1.0f - sP[r]) + i;
            float s = (v >= THRESH) ? 1.0f : 0.0f;
            iS[r] = i; vS[r] = v; sP[r] = s;
            const int n = n0 + r + 8 * half, o = p0 + col;
            if (t + 1 < TSTEPS)
                out[((size_t)n * COUT + o) * TSTEPS + t + 1] = s;  // delay shift
            swr[(size_t)n * COUT + o] = (v >= THRESH) ? (unsigned short)0x3F80
                                                      : (unsigned short)0;
        }

        // prefetch next step's z tile while we wait at the barrier
        if (t + 1 < TSTEPS)
            __builtin_prefetch(z + ((size_t)(t + 1) * NB + n0 + m) * COUT + p0, 0, 3);

        // ---- grid-wide barrier (monotonic counter, 8 co-resident WGs) ----
        __syncthreads();
        if (threadIdx.x == 0) {
            __hip_atomic_fetch_add(counter, 1u, __ATOMIC_RELEASE, __HIP_MEMORY_SCOPE_AGENT);
            const unsigned target = (unsigned)(t + 1) * SCAN_WGS;
            while (__hip_atomic_load(counter, __ATOMIC_ACQUIRE,
                                     __HIP_MEMORY_SCOPE_AGENT) < target)
                __builtin_amdgcn_s_sleep(2);
        }
        __syncthreads();
    }
}

// ---------------------------------------------------------------------------
// Launch: inputs = { x:[32,512,1024] f32, W:[512,512] f32, R:[512,512] f32 }
// out = [32,512,1024] f32. ws: [counter | sbuf 64KB | z 64MB]
// ---------------------------------------------------------------------------
extern "C" void kernel_launch(void* const* d_in, const int* in_sizes, int n_in,
                              void* d_out, int out_size, void* d_ws, size_t ws_size,
                              hipStream_t stream) {
    const float* x = (const float*)d_in[0];
    const float* W = (const float*)d_in[1];
    const float* R = (const float*)d_in[2];
    float* out = (float*)d_out;

    char* ws = (char*)d_ws;
    unsigned*       counter = (unsigned*)ws;
    unsigned short* sbuf    = (unsigned short*)(ws + 256);
    float*          z       = (float*)(ws + 256 + (size_t)2 * NB * COUT * sizeof(unsigned short));

    lif_init<<<(2 * NB * COUT + 255) / 256, 256, 0, stream>>>(sbuf, counter);

    // 64 t-tiles * 32 n * 32 o-tiles = 65536 tiles, 8 waves per 256-thread block
    input_gemm<<<(TSTEPS / 16) * NB * (COUT / 16) / 8, 256, 0, stream>>>(x, W, z);

    lif_scan<<<SCAN_WGS, 256, 0, stream>>>(z, R, out, sbuf, counter);
}
